// Resnet18_19121194401914
// MI455X (gfx1250) — matrix operand order repaired and checked by reading, not compiled, not run on hardware
//
#include <hip/hip_runtime.h>
#include <hip/hip_bf16.h>

// ---------------------------------------------------------------------------
// PVNet-style network on MI455X (gfx1250, wave32).
// Heavy convs -> implicit-GEMM with v_wmma_f32_16x16x32_f16, f32 accumulate.
// ---------------------------------------------------------------------------

typedef __attribute__((ext_vector_type(16))) _Float16 v16h;
typedef __attribute__((ext_vector_type(8)))  _Float16 v8h;
typedef __attribute__((ext_vector_type(8)))  float    v8f;

#define BB    4
#define IH    480
#define IW    640
#define NPTS  200
#define VN    9          // VER_DIM/2
#define NPNP  (NPTS*VN)  // 1800
#define NEG_INF (-__builtin_huge_valf())

// ===========================================================================
// Implicit-GEMM conv + fused BN/bias + residual + activation.
//   M = Cout, N = Bn*Hout*Wout, K = Cin*KH*KW
//   Workgroup: 256 threads = 8 waves, 2(M) x 4(N); wave tile 32x32.
//   LDS tiles: A[64][32] f16, B[128][32] f16 (both [row][K] for b128 frag loads)
// actMode: 0=none 1=relu 2=leaky(0.1);  bnMode: 1=BN(g,b,m,v) 0=bias-only(g)
// ===========================================================================
__global__ __launch_bounds__(256)
void conv_wmma_kernel(const float* __restrict__ X, const float* __restrict__ Wt,
                      const float* __restrict__ bnG, const float* __restrict__ bnB,
                      const float* __restrict__ bnM, const float* __restrict__ bnV,
                      const float* __restrict__ Res, float* __restrict__ Out,
                      int Bn, int Cin, int Hin, int Win,
                      int Cout, int Hout, int Wout,
                      int KH, int KW, int stride, int pad, int dil,
                      int bnMode, int actMode)
{
    __shared__ __align__(16) _Float16 As[64 * 32];
    __shared__ __align__(16) _Float16 Bs[128 * 32];

    const int P  = Bn * Hout * Wout;
    const int K  = Cin * KH * KW;
    const int HW = Hout * Wout;

    const int pBase = blockIdx.x * 128;
    const int mBase = blockIdx.y * 64;

    const int tid   = threadIdx.x;
    const int lane  = tid & 31;
    const int wave  = tid >> 5;
    const int waveM = wave & 1;   // 2 tiles of 32 rows
    const int waveN = wave >> 1;  // 4 tiles of 32 cols

    v8f acc[2][2];
#pragma unroll
    for (int i = 0; i < 2; ++i)
#pragma unroll
        for (int j = 0; j < 2; ++j)
            acc[i][j] = (v8f){0.f,0.f,0.f,0.f,0.f,0.f,0.f,0.f};

    // Pre-decode the output position this thread stages B for.
    const int bn    = tid >> 1;          // 0..127 : tile column
    const int bkOff = (tid & 1) * 16;    // 16 K-values per staging thread
    const int bp    = pBase + bn;
    int bImg = 0, bOh = 0, bOw = 0;
    const bool bValid = bp < P;
    if (bValid) {
        bImg = bp / HW;
        int r = bp - bImg * HW;
        bOh = r / Wout;
        bOw = r - bOh * Wout;
    }
    const int aRow = tid >> 2;           // 0..63 : tile row for A staging
    const int aOff = (tid & 3) * 8;
    const int aM   = mBase + aRow;

    for (int k0 = 0; k0 < K; k0 += 32) {
        // ---- stage A (weights, f32 -> f16), [m][k] ----
#pragma unroll
        for (int j = 0; j < 8; ++j) {
            int k = k0 + aOff + j;
            float v = 0.f;
            if (aM < Cout && k < K) v = Wt[(long long)aM * K + k];
            As[aRow * 32 + aOff + j] = (_Float16)v;
        }
        // prefetch next weight tile (global_prefetch_b8)
        if (aM < Cout && k0 + 32 < K)
            __builtin_prefetch(&Wt[(long long)aM * K + k0 + 32 + aOff], 0, 1);

        // ---- stage B (im2col gather, f32 -> f16), [n][k] ----
#pragma unroll 4
        for (int j = 0; j < 16; ++j) {
            int k = k0 + bkOff + j;
            float v = 0.f;
            if (bValid && k < K) {
                int ci  = k / (KH * KW);
                int r2  = k - ci * KH * KW;
                int kh  = r2 / KW;
                int kw2 = r2 - kh * KW;
                int ih  = bOh * stride - pad + kh * dil;
                int iw  = bOw * stride - pad + kw2 * dil;
                if (ih >= 0 && ih < Hin && iw >= 0 && iw < Win)
                    v = X[(((long long)bImg * Cin + ci) * Hin + ih) * Win + iw];
            }
            Bs[bn * 32 + bkOff + j] = (_Float16)v;
        }
        __syncthreads();

        // ---- fragments per documented 16-bit layouts ----
        // A 16x32: lanes 0-15 row M=lane, K chunks {0..7,16..23};
        //          lanes 16-31 row M=lane-16, K chunks {8..15,24..31}
        const int kbA = (lane >> 4) * 8;
        v16h afrag[2];
#pragma unroll
        for (int tm = 0; tm < 2; ++tm) {
            int mrow = waveM * 32 + tm * 16 + (lane & 15);
            v8h a0 = *(const v8h*)&As[mrow * 32 + kbA];
            v8h a1 = *(const v8h*)&As[mrow * 32 + kbA + 16];
            afrag[tm] = __builtin_shufflevector(a0, a1,
                0,1,2,3,4,5,6,7,8,9,10,11,12,13,14,15);
        }
        // B 32x16: lanes 0-15 col N=lane, K chunks {0..7,16..23}; lanes 16-31 col N=lane-16, K chunks {8..15,24..31}
        const int kbB = (lane >> 4) * 8;
        v16h bfrag[2];
#pragma unroll
        for (int tn = 0; tn < 2; ++tn) {
            int ncol = waveN * 32 + tn * 16 + (lane & 15);
            v8h b0 = *(const v8h*)&Bs[ncol * 32 + kbB];
            v8h b1 = *(const v8h*)&Bs[ncol * 32 + kbB + 16];
            bfrag[tn] = __builtin_shufflevector(b0, b1,
                0,1,2,3,4,5,6,7,8,9,10,11,12,13,14,15);
        }

#pragma unroll
        for (int tm = 0; tm < 2; ++tm)
#pragma unroll
            for (int tn = 0; tn < 2; ++tn)
                acc[tm][tn] = __builtin_amdgcn_wmma_f32_16x16x32_f16(
                    false, afrag[tm], false, bfrag[tn],
                    (short)0, acc[tm][tn], false, false);

        __syncthreads();
    }

    // ---- epilogue: BN/bias + residual + activation ----
    const int rowHalf = (lane >> 4) * 8;  // C/D layout: VGPR i -> M=i (lo) / 8+i (hi)
#pragma unroll
    for (int tn = 0; tn < 2; ++tn) {
        int col = pBase + waveN * 32 + tn * 16 + (lane & 15);
        if (col >= P) continue;
        int img = col / HW;
        int sp  = col - img * HW;
#pragma unroll
        for (int tm = 0; tm < 2; ++tm) {
#pragma unroll
            for (int i = 0; i < 8; ++i) {
                int row = mBase + waveM * 32 + tm * 16 + rowHalf + i;
                if (row >= Cout) continue;
                float s, bi;
                if (bnMode) {
                    float g = bnG[row], b = bnB[row], m = bnM[row], v = bnV[row];
                    s  = g * rsqrtf(v + 1e-5f);
                    bi = b - m * s;
                } else {
                    s  = 1.f;
                    bi = bnG ? bnG[row] : 0.f;
                }
                float val = acc[tm][tn][i] * s + bi;
                long long o = ((long long)img * Cout + row) * HW + sp;
                if (Res) val += Res[o];
                if (actMode == 1)      val = fmaxf(val, 0.f);
                else if (actMode == 2) val = val > 0.f ? val : 0.1f * val;
                Out[o] = val;
            }
        }
    }
}

// ===========================================================================
// Support kernels
// ===========================================================================
__global__ void maxpool_kernel(const float* __restrict__ in, float* __restrict__ out,
                               int BC, int Hin, int Win, int Hout, int Wout)
{
    long long gid = (long long)blockIdx.x * blockDim.x + threadIdx.x;
    long long tot = (long long)BC * Hout * Wout;
    if (gid >= tot) return;
    int bc = (int)(gid / (Hout * Wout));
    int r  = (int)(gid - (long long)bc * Hout * Wout);
    int oh = r / Wout, ow = r - (r / Wout) * Wout;
    const float* p = in + (long long)bc * Hin * Win;
    float m = NEG_INF;
    for (int dy = 0; dy < 3; ++dy)
        for (int dx = 0; dx < 3; ++dx) {
            int ih = oh * 2 - 1 + dy, iw = ow * 2 - 1 + dx;
            if (ih >= 0 && ih < Hin && iw >= 0 && iw < Win)
                m = fmaxf(m, p[ih * Win + iw]);
        }
    out[gid] = m;
}

// 2x resize matching the reference's src = arange(2n)*(n-1)/(2n-1) bilinear
__global__ void up2x_kernel(const float* __restrict__ in, float* __restrict__ out,
                            int BC, int Hin, int Win)
{
    int Hout = 2 * Hin, Wout = 2 * Win;
    long long gid = (long long)blockIdx.x * blockDim.x + threadIdx.x;
    long long tot = (long long)BC * Hout * Wout;
    if (gid >= tot) return;
    int bc = (int)(gid / ((long long)Hout * Wout));
    int r  = (int)(gid - (long long)bc * Hout * Wout);
    int oh = r / Wout, ow = r - (r / Wout) * Wout;
    float sy = (float)oh * (float)(Hin - 1) / (float)(Hout - 1);
    float sx = (float)ow * (float)(Win - 1) / (float)(Wout - 1);
    int y0 = (int)floorf(sy); int y1 = min(y0 + 1, Hin - 1); float ty = sy - y0;
    int x0 = (int)floorf(sx); int x1 = min(x0 + 1, Win - 1); float tx = sx - x0;
    const float* p = in + (long long)bc * Hin * Win;
    float v = (p[y0 * Win + x0] * (1.f - tx) + p[y0 * Win + x1] * tx) * (1.f - ty)
            + (p[y1 * Win + x0] * (1.f - tx) + p[y1 * Win + x1] * tx) * ty;
    out[gid] = v;
}

// copy src (Bn,Cs,HW) -> dst (Bn,Cdst,HW) at channel offset cOff
__global__ void concat_copy_kernel(const float* __restrict__ src, float* __restrict__ dst,
                                   int Bn, int Cs, int Cdst, int cOff, int HW)
{
    long long gid = (long long)blockIdx.x * blockDim.x + threadIdx.x;
    long long tot = (long long)Bn * Cs * HW;
    if (gid >= tot) return;
    int b = (int)(gid / ((long long)Cs * HW));
    long long r = gid - (long long)b * Cs * HW;
    int c = (int)(r / HW);
    int p = (int)(r - (long long)c * HW);
    dst[((long long)b * Cdst + cOff + c) * HW + p] = src[gid];
}

// headBuf (B,20,HW) -> seg (B,2,HW) + ver (B,18,HW) regions of d_out
__global__ void split_head_kernel(const float* __restrict__ h,
                                  float* __restrict__ seg, float* __restrict__ ver, int HW)
{
    long long gid = (long long)blockIdx.x * blockDim.x + threadIdx.x;
    long long tot = (long long)BB * 20 * HW;
    if (gid >= tot) return;
    int b = (int)(gid / ((long long)20 * HW));
    long long r = gid - (long long)b * 20 * HW;
    int c = (int)(r / HW);
    int p = (int)(r - (long long)c * HW);
    float v = h[gid];
    if (c < 2) seg[((long long)b * 2 + c) * HW + p] = v;
    else       ver[((long long)b * 18 + (c - 2)) * HW + p] = v;
}

__global__ void pri_kernel(const int* __restrict__ seg, const float* __restrict__ noise,
                           float* __restrict__ pri, long long n)
{
    long long gid = (long long)blockIdx.x * blockDim.x + threadIdx.x;
    if (gid >= n) return;
    pri[gid] = (seg[gid] > 0) ? noise[gid] : -1.f;
}

// iterative block-wide argmax top-k (stable: ties -> lower index)
__global__ void topk_kernel(float* __restrict__ pri, int* __restrict__ idxOut, int HW)
{
    int b = blockIdx.x;
    float* p = pri + (long long)b * HW;
    __shared__ float sv[256];
    __shared__ int   si[256];
    for (int t = 0; t < NPTS; ++t) {
        float best = NEG_INF; int bi = 0x7fffffff;
        for (int i = threadIdx.x; i < HW; i += 256) {
            float v = p[i];
            if (v > best || (v == best && i < bi)) { best = v; bi = i; }
        }
        sv[threadIdx.x] = best; si[threadIdx.x] = bi;
        __syncthreads();
        for (int s = 128; s > 0; s >>= 1) {
            if (threadIdx.x < s) {
                if (sv[threadIdx.x + s] > sv[threadIdx.x] ||
                    (sv[threadIdx.x + s] == sv[threadIdx.x] && si[threadIdx.x + s] < si[threadIdx.x])) {
                    sv[threadIdx.x] = sv[threadIdx.x + s];
                    si[threadIdx.x] = si[threadIdx.x + s];
                }
            }
            __syncthreads();
        }
        if (threadIdx.x == 0) { idxOut[b * NPTS + t] = si[0]; p[si[0]] = NEG_INF; }
        __syncthreads();
    }
}

__global__ void bbox_kernel(const int* __restrict__ seg, int* __restrict__ bbox, int H, int W)
{
    int b = blockIdx.x;
    __shared__ int s[4];
    if (threadIdx.x == 0) { s[0] = H; s[1] = -1; s[2] = W; s[3] = -1; }
    __syncthreads();
    int rmin = H, rmax = -1, cmin = W, cmax = -1;
    long long HWll = (long long)H * W;
    for (long long i = threadIdx.x; i < HWll; i += blockDim.x) {
        if (seg[(long long)b * HWll + i] > 0) {
            int r = (int)(i / W), c = (int)(i - (i / W) * W);
            rmin = min(rmin, r); rmax = max(rmax, r);
            cmin = min(cmin, c); cmax = max(cmax, c);
        }
    }
    atomicMin(&s[0], rmin); atomicMax(&s[1], rmax);
    atomicMin(&s[2], cmin); atomicMax(&s[3], cmax);
    __syncthreads();
    if (threadIdx.x < 4) bbox[b * 4 + threadIdx.x] = s[threadIdx.x];
}

// 256 threads = 64 channels x 4 cells; feat[b][ch*4+cell]
__global__ void cellfeat_kernel(const float* __restrict__ fm, const int* __restrict__ bbox,
                                float* __restrict__ feat, int C, int H4, int W4)
{
    int b    = blockIdx.x;
    int ch   = threadIdx.x >> 2;
    int cell = threadIdx.x & 3;
    int i = cell >> 1, j = cell & 1;
    int rmin = bbox[b*4+0], rmax = bbox[b*4+1], cmin = bbox[b*4+2], cmax = bbox[b*4+3];
    int b0 = rmin / 4, b1 = cmin / 4;
    int b2 = max(b0 + 1, rmax / 4), b3 = max(b1 + 1, cmax / 4);
    int kh = (b2 - b0) / 2, kw = (b3 - b1) / 2;
    int r0 = max(b0 + i * kh, 0), r1 = min(b0 + (i + 1) * kh, H4);
    int c0 = max(b1 + j * kw, 0), c1 = min(b1 + (j + 1) * kw, W4);
    float m = NEG_INF;
    const float* base = fm + ((long long)b * C + ch) * H4 * W4;
    for (int r = r0; r < r1; ++r)
        for (int c = c0; c < c1; ++c)
            m = fmaxf(m, base[r * W4 + c]);
    feat[(long long)b * C * 4 + ch * 4 + cell] = m;
}

// build pts (B,4,1800) from sampled indices + ver_pred (B,18,H,W)
__global__ void pts_kernel(const int* __restrict__ idx, const float* __restrict__ ver,
                           float* __restrict__ pts, int H, int W)
{
    int gid = blockIdx.x * blockDim.x + threadIdx.x;
    int tot = BB * NPNP;
    if (gid >= tot) return;
    int b  = gid / NPNP;
    int r0 = gid - b * NPNP;
    int pt = r0 / VN, j = r0 - pt * VN;
    int id = idx[b * NPTS + pt];
    int r = id / W, c = id - (id / W) * W;
    long long HWll = (long long)H * W;
    long long vb = (long long)b * 18 * HWll;
    float dx = ver[vb + (2 * j)     * HWll + (long long)r * W + c];
    float dy = ver[vb + (2 * j + 1) * HWll + (long long)r * W + c];
    int n = pt * VN + j;
    long long pb = (long long)b * 4 * NPNP;
    pts[pb + 0 * NPNP + n] = (float)c / (float)W - 0.5f;
    pts[pb + 1 * NPNP + n] = (float)r / (float)H - 0.5f;
    pts[pb + 2 * NPNP + n] = dx;
    pts[pb + 3 * NPNP + n] = dy;
}

// out[b][o][n] = act(sum_f w[o][f]*in[b][f][n] + bias[o])
__global__ void lin1d_kernel(const float* __restrict__ w, const float* __restrict__ bias,
                             const float* __restrict__ in, float* __restrict__ out,
                             int Bn, int O, int F, int N, int act)
{
    long long gid = (long long)blockIdx.x * blockDim.x + threadIdx.x;
    long long tot = (long long)Bn * O * N;
    if (gid >= tot) return;
    int b = (int)(gid / ((long long)O * N));
    long long r = gid - (long long)b * O * N;
    int o = (int)(r / N);
    int n = (int)(r - (long long)o * N);
    float s = bias[o];
    const float* wi = w + (long long)o * F;
    const float* hi = in + (long long)b * F * N + n;
    for (int f = 0; f < F; ++f) s += wi[f] * hi[(long long)f * N];
    if (act) s = fmaxf(s, 0.f);
    out[((long long)b * O + o) * N + n] = s;
}

__global__ void maxn_kernel(const float* __restrict__ in, float* __restrict__ out,
                            int Bn, int O, int N)
{
    int gid = blockIdx.x * blockDim.x + threadIdx.x;
    if (gid >= Bn * O) return;
    const float* p = in + (long long)gid * N;
    float m = NEG_INF;
    for (int n = 0; n < N; ++n) m = fmaxf(m, p[n]);
    out[gid] = m;
}

__global__ void veccat_kernel(const float* __restrict__ a, const float* __restrict__ b,
                              float* __restrict__ out, int Bn, int Ca, int Cb)
{
    int gid = blockIdx.x * blockDim.x + threadIdx.x;
    int C = Ca + Cb;
    if (gid >= Bn * C) return;
    int bi = gid / C, c = gid - bi * C;
    out[gid] = (c < Ca) ? a[bi * Ca + c] : b[bi * Cb + (c - Ca)];
}

__global__ void fc_kernel(const float* __restrict__ w, const float* __restrict__ bias,
                          const float* __restrict__ in, float* __restrict__ out,
                          int Bn, int O, int F, int act)
{
    int gid = blockIdx.x * blockDim.x + threadIdx.x;
    if (gid >= Bn * O) return;
    int b = gid / O, o = gid - b * O;
    float s = bias[o];
    const float* wi = w + (long long)o * F;
    const float* hi = in + (long long)b * F;
    for (int f = 0; f < F; ++f) s += wi[f] * hi[f];
    if (act) s = fmaxf(s, 0.f);
    out[gid] = s;
}

__global__ void pose_kernel(const float* __restrict__ pose,
                            float* __restrict__ poseOut, float* __restrict__ maskOut)
{
    int b = threadIdx.x;
    if (b >= BB) return;
    const float* p = pose + b * 7;
    float w = p[0], x = p[1], y = p[2], z = p[3];
    float n = sqrtf(w*w + x*x + y*y + z*z);
    w /= n; x /= n; y /= n; z /= n;
    float R[3][3] = {
        {1.f - 2.f*(y*y + z*z), 2.f*(x*y - z*w),       2.f*(x*z + y*w)},
        {2.f*(x*y + z*w),       1.f - 2.f*(x*x + z*z), 2.f*(y*z - x*w)},
        {2.f*(x*z - y*w),       2.f*(y*z + x*w),       1.f - 2.f*(x*x + y*y)}};
    for (int i = 0; i < 3; ++i) {
        for (int j = 0; j < 3; ++j) poseOut[b*12 + i*4 + j] = R[i][j];
        poseOut[b*12 + i*4 + 3] = p[4 + i];
    }
    maskOut[b] = 1.f;
}

// ===========================================================================
// Host orchestration
// ===========================================================================
static inline int odim(int n, int k, int s, int p, int d) {
    return (n + 2 * p - d * (k - 1) - 1) / s + 1;
}

struct BnPtrs { const float *g, *b, *m, *v; };
static inline BnPtrs bnAt(void* const* din, int base) {
    // bnp dict {g,b,m,v} flattened alphabetically: b,g,m,v
    return { (const float*)din[base + 1], (const float*)din[base + 0],
             (const float*)din[base + 2], (const float*)din[base + 3] };
}

static void conv(hipStream_t st, const float* X, const float* Wt, BnPtrs bn, int bnMode,
                 const float* res, float* out,
                 int Bn, int Cin, int Hin, int Win, int Cout,
                 int k, int stride, int pad, int dil, int act)
{
    int Hout = odim(Hin, k, stride, pad, dil);
    int Wout = odim(Win, k, stride, pad, dil);
    long long P = (long long)Bn * Hout * Wout;
    dim3 grid((unsigned)((P + 127) / 128), (unsigned)((Cout + 63) / 64));
    conv_wmma_kernel<<<grid, 256, 0, st>>>(X, Wt, bn.g, bn.b, bn.m, bn.v, res, out,
                                           Bn, Cin, Hin, Win, Cout, Hout, Wout,
                                           k, k, stride, pad, dil, bnMode, act);
}

struct BlockIdx { int c1, c2, cd, n1, n2, nd; };

static void resblock(hipStream_t st, void* const* din, const float* in,
                     float* t1, float* scbuf, float* out, BlockIdx bi,
                     int Bn, int Cin, int Hin, int Win, int Cout, int stride, int dil)
{
    conv(st, in, (const float*)din[bi.c1], bnAt(din, bi.n1), 1, nullptr, t1,
         Bn, Cin, Hin, Win, Cout, 3, stride, dil, dil, /*relu*/1);
    const float* res;
    if (bi.cd >= 0) {
        conv(st, in, (const float*)din[bi.cd], bnAt(din, bi.nd), 1, nullptr, scbuf,
             Bn, Cin, Hin, Win, Cout, 1, stride, 0, 1, /*none*/0);
        res = scbuf;
    } else {
        res = in;
    }
    int H2 = odim(Hin, 3, stride, dil, dil);
    int W2 = odim(Win, 3, stride, dil, dil);
    conv(st, t1, (const float*)din[bi.c2], bnAt(din, bi.n2), 1, res, out,
         Bn, Cout, H2, W2, Cout, 3, 1, dil, dil, /*relu*/1);
}

extern "C" void kernel_launch(void* const* d_in, const int* in_sizes, int n_in,
                              void* d_out, int out_size, void* d_ws, size_t ws_size,
                              hipStream_t stream)
{
    (void)in_sizes; (void)n_in; (void)out_size; (void)ws_size;

    const float* x      = (const float*)d_in[0];
    const int*   seg_gt = (const int*)d_in[1];
    const float* noise  = (const float*)d_in[2];

    // ---- param leaf indices (jax tree_flatten: sorted dict keys, base = 3) ----
    // bn1=3..6, c2_bn=7..10, c2_w=11, c4_bn=12..15, c4_w=16, c8_bn=17..20,
    // c8_w=21, conv1=22, fc_bn=23..26, fc_w=27, head_b=28, head_w=29,
    // layer1..4 = 30..124, pnp = 125..136, raw_bn=137..140, raw_w=141
    const int I_bn1 = 3, I_c2bn = 7, I_c2w = 11, I_c4bn = 12, I_c4w = 16;
    const int I_c8bn = 17, I_c8w = 21, I_conv1 = 22, I_fcbn = 23, I_fcw = 27;
    const int I_headb = 28, I_headw = 29, I_rawbn = 137, I_raww = 141;
    const BlockIdx L1B0 = {30, 31, -1, 32, 36, -1};
    const BlockIdx L1B1 = {40, 41, -1, 42, 46, -1};
    const BlockIdx L2B0 = {50, 51, 52, 53, 57, 61};
    const BlockIdx L2B1 = {65, 66, -1, 67, 71, -1};
    const BlockIdx L3B0 = {75, 76, 77, 78, 82, 86};
    const BlockIdx L3B1 = {90, 91, -1, 92, 96, -1};
    const BlockIdx L4B0 = {100, 101, 102, 103, 107, 111};
    const BlockIdx L4B1 = {115, 116, -1, 117, 121, -1};
    // pnp (sorted: f1,f2,f3,p1,p2,p3; lin sorted: b,w)
    const int I_f1b = 125, I_f1w = 126, I_f2b = 127, I_f2w = 128, I_f3b = 129, I_f3w = 130;
    const int I_p1b = 131, I_p1w = 132, I_p2b = 133, I_p2w = 134, I_p3b = 135, I_p3w = 136;
#define PF(i) ((const float*)d_in[(i)])

    // ---- workspace bump allocator ----
    char* wsBase = (char*)d_ws;
    size_t wsOff = 0;
    auto allocf = [&](long long elems) -> float* {
        float* p = (float*)(wsBase + wsOff);
        wsOff += (size_t)(((elems * 4 + 255) / 256) * 256);
        return p;
    };

    float* x2s    = allocf((long long)BB * 64 * 240 * 320);
    float* pooled = allocf((long long)BB * 64 * 120 * 160);
    float* t64a   = allocf((long long)BB * 64 * 120 * 160);
    float* t64b   = allocf((long long)BB * 64 * 120 * 160);
    float* x4s    = allocf((long long)BB * 64 * 120 * 160);
    float* x8s    = allocf((long long)BB * 128 * 60 * 80);
    float* qa     = allocf((long long)BB * 512 * 60 * 80);
    float* qb     = allocf((long long)BB * 512 * 60 * 80);
    float* qc     = allocf((long long)BB * 512 * 60 * 80);
    float* qd     = allocf((long long)BB * 512 * 60 * 80);
    float* fm4    = allocf((long long)BB * 64 * 120 * 160);   // fm_copy
    float* bigA   = allocf((long long)BB * 35 * 480 * 640);
    float* bigB   = allocf((long long)BB * 32 * 480 * 640);
    float* headB  = allocf((long long)BB * 20 * 480 * 640);
    float* pri    = allocf((long long)BB * IH * IW);
    int*   idxbuf = (int*)allocf(BB * NPTS);
    int*   bboxb  = (int*)allocf(BB * 4);
    float* featb  = allocf(BB * 256);
    float* ptsb   = allocf((long long)BB * 4 * NPNP);
    float* h1a    = allocf((long long)BB * 256 * NPNP);
    float* h1b    = allocf((long long)BB * 128 * NPNP);
    float* gmax   = allocf(BB * 256);
    float* g512   = allocf(BB * 512);
    float* g256   = allocf(BB * 256);
    float* g128   = allocf(BB * 128);
    float* pose7  = allocf(BB * 7);

    const long long HWfull = (long long)IH * IW;
    float* segOut  = (float*)d_out;                       // (B,2,H,W)
    float* verOut  = segOut + (long long)BB * 2 * HWfull; // (B,18,H,W)
    float* poseOut = verOut + (long long)BB * 18 * HWfull;
    float* maskOut = poseOut + BB * 12;

    // =================== Encoder ===================
    conv(stream, x, PF(I_conv1), bnAt(d_in, I_bn1), 1, nullptr, x2s,
         BB, 3, IH, IW, 64, 7, 2, 3, 1, 1);                        // -> 64x240x320
    {
        long long tot = (long long)BB * 64 * 120 * 160;
        maxpool_kernel<<<(unsigned)((tot + 255) / 256), 256, 0, stream>>>(
            x2s, pooled, BB * 64, 240, 320, 120, 160);
    }
    resblock(stream, d_in, pooled, t64a, nullptr, t64b, L1B0, BB, 64, 120, 160, 64, 1, 1);
    resblock(stream, d_in, t64b,   t64a, nullptr, x4s,  L1B1, BB, 64, 120, 160, 64, 1, 1);
    resblock(stream, d_in, x4s, qa, qb, qc,  L2B0, BB, 64, 120, 160, 128, 2, 1);
    resblock(stream, d_in, qc,  qa, nullptr, x8s, L2B1, BB, 128, 60, 80, 128, 1, 1);
    resblock(stream, d_in, x8s, qa, qb, qc,  L3B0, BB, 128, 60, 80, 256, 1, 2);
    resblock(stream, d_in, qc,  qa, nullptr, qb,  L3B1, BB, 256, 60, 80, 256, 1, 2);
    resblock(stream, d_in, qb,  qa, qc, qd,  L4B0, BB, 256, 60, 80, 512, 1, 4);
    resblock(stream, d_in, qd,  qa, nullptr, qb,  L4B1, BB, 512, 60, 80, 512, 1, 4);

    // =================== Decoder ===================
    conv(stream, qb, PF(I_fcw), bnAt(d_in, I_fcbn), 1, nullptr, qa,
         BB, 512, 60, 80, 256, 3, 1, 1, 1, 1);                     // xfc 256x60x80
    {
        int HW8 = 60 * 80;
        long long tot = (long long)BB * 256 * HW8;
        concat_copy_kernel<<<(unsigned)((tot + 255) / 256), 256, 0, stream>>>(
            qa, qc, BB, 256, 384, 0, HW8);
        tot = (long long)BB * 128 * HW8;
        concat_copy_kernel<<<(unsigned)((tot + 255) / 256), 256, 0, stream>>>(
            x8s, qc, BB, 128, 384, 256, HW8);
    }
    conv(stream, qc, PF(I_c8w), bnAt(d_in, I_c8bn), 1, nullptr, qd,
         BB, 384, 60, 80, 128, 3, 1, 1, 1, 2);                     // fm 128x60x80
    {
        long long tot = (long long)BB * 128 * 120 * 160;
        up2x_kernel<<<(unsigned)((tot + 255) / 256), 256, 0, stream>>>(qd, qb, BB * 128, 60, 80);
    }
    {
        int HW4 = 120 * 160;
        long long tot = (long long)BB * 128 * HW4;
        concat_copy_kernel<<<(unsigned)((tot + 255) / 256), 256, 0, stream>>>(
            qb, bigA, BB, 128, 192, 0, HW4);
        tot = (long long)BB * 64 * HW4;
        concat_copy_kernel<<<(unsigned)((tot + 255) / 256), 256, 0, stream>>>(
            x4s, bigA, BB, 64, 192, 128, HW4);
    }
    conv(stream, bigA, PF(I_c4w), bnAt(d_in, I_c4bn), 1, nullptr, fm4,
         BB, 192, 120, 160, 64, 3, 1, 1, 1, 2);                    // fm_copy 64x120x160
    {
        long long tot = (long long)BB * 64 * 240 * 320;
        up2x_kernel<<<(unsigned)((tot + 255) / 256), 256, 0, stream>>>(fm4, bigB, BB * 64, 120, 160);
    }
    {
        int HW2 = 240 * 320;
        long long tot = (long long)BB * 64 * HW2;
        concat_copy_kernel<<<(unsigned)((tot + 255) / 256), 256, 0, stream>>>(
            bigB, bigA, BB, 64, 128, 0, HW2);
        concat_copy_kernel<<<(unsigned)((tot + 255) / 256), 256, 0, stream>>>(
            x2s, bigA, BB, 64, 128, 64, HW2);
    }
    conv(stream, bigA, PF(I_c2w), bnAt(d_in, I_c2bn), 1, nullptr, qa,
         BB, 128, 240, 320, 32, 3, 1, 1, 1, 2);                    // fm 32x240x320
    {
        long long tot = (long long)BB * 32 * HWfull;
        up2x_kernel<<<(unsigned)((tot + 255) / 256), 256, 0, stream>>>(qa, bigB, BB * 32, 240, 320);
    }
    {
        long long tot = (long long)BB * 32 * HWfull;
        concat_copy_kernel<<<(unsigned)((tot + 255) / 256), 256, 0, stream>>>(
            bigB, bigA, BB, 32, 35, 0, (int)HWfull);
        tot = (long long)BB * 3 * HWfull;
        concat_copy_kernel<<<(unsigned)((tot + 255) / 256), 256, 0, stream>>>(
            x, bigA, BB, 3, 35, 32, (int)HWfull);
    }
    conv(stream, bigA, PF(I_raww), bnAt(d_in, I_rawbn), 1, nullptr, bigB,
         BB, 35, IH, IW, 32, 3, 1, 1, 1, 2);                       // raw 32x480x640
    BnPtrs headBn = { PF(I_headb), nullptr, nullptr, nullptr };
    conv(stream, bigB, PF(I_headw), headBn, 0, nullptr, headB,
         BB, 32, IH, IW, 20, 1, 1, 0, 1, 0);                       // head 20x480x640
    {
        long long tot = (long long)BB * 20 * HWfull;
        split_head_kernel<<<(unsigned)((tot + 255) / 256), 256, 0, stream>>>(
            headB, segOut, verOut, (int)HWfull);
    }

    // =================== Sampling + PnP ===================
    {
        long long tot = (long long)BB * HWfull;
        pri_kernel<<<(unsigned)((tot + 255) / 256), 256, 0, stream>>>(seg_gt, noise, pri, tot);
    }
    topk_kernel<<<BB, 256, 0, stream>>>(pri, idxbuf, (int)HWfull);
    bbox_kernel<<<BB, 256, 0, stream>>>(seg_gt, bboxb, IH, IW);
    cellfeat_kernel<<<BB, 256, 0, stream>>>(fm4, bboxb, featb, 64, 120, 160);
    pts_kernel<<<(BB * NPNP + 255) / 256, 256, 0, stream>>>(idxbuf, verOut, ptsb, IH, IW);

    lin1d_kernel<<<(unsigned)(((long long)BB * 64 * NPNP + 255) / 256), 256, 0, stream>>>(
        PF(I_p1w), PF(I_p1b), ptsb, h1a, BB, 64, 4, NPNP, 1);
    lin1d_kernel<<<(unsigned)(((long long)BB * 128 * NPNP + 255) / 256), 256, 0, stream>>>(
        PF(I_p2w), PF(I_p2b), h1a, h1b, BB, 128, 64, NPNP, 1);
    lin1d_kernel<<<(unsigned)(((long long)BB * 256 * NPNP + 255) / 256), 256, 0, stream>>>(
        PF(I_p3w), PF(I_p3b), h1b, h1a, BB, 256, 128, NPNP, 1);
    maxn_kernel<<<(BB * 256 + 255) / 256, 256, 0, stream>>>(h1a, gmax, BB, 256, NPNP);
    veccat_kernel<<<(BB * 512 + 255) / 256, 256, 0, stream>>>(gmax, featb, g512, BB, 256, 256);
    fc_kernel<<<(BB * 256 + 255) / 256, 256, 0, stream>>>(PF(I_f1w), PF(I_f1b), g512, g256, BB, 256, 512, 1);
    fc_kernel<<<(BB * 128 + 255) / 256, 256, 0, stream>>>(PF(I_f2w), PF(I_f2b), g256, g128, BB, 128, 256, 1);
    fc_kernel<<<1, 64, 0, stream>>>(PF(I_f3w), PF(I_f3b), g128, pose7, BB, 7, 128, 0);
    pose_kernel<<<1, 32, 0, stream>>>(pose7, poseOut, maskOut);
}
